// Asymm_3d_spconv_31447750541871
// MI455X (gfx1250) — matrix operand order repaired and checked by reading, not compiled, not run on hardware
//
#include <hip/hip_runtime.h>

#define N_VOX  200000
#define COUTC  64
#define KOFF   9
#define BN_EPS 1e-5f

typedef __attribute__((ext_vector_type(16))) __bf16 v16bf;
typedef __attribute__((ext_vector_type(8)))  __bf16 v8bf;
typedef __attribute__((ext_vector_type(8)))  float  v8f;
typedef int v4i_ __attribute__((vector_size(16)));   // matches builtin's V4i

union AFrag { v16bf v; v8bf h[2]; };

#if defined(__AMDGCN__) && \
    __has_builtin(__builtin_amdgcn_global_load_async_to_lds_b128) && \
    __has_builtin(__builtin_amdgcn_s_wait_asynccnt)
#define USE_ASYNC_LDS 1
#else
#define USE_ASYNC_LDS 0
#endif

static __device__ __forceinline__ unsigned short f2bf_raw(float f) {
    union { float f; unsigned int u; } x; x.f = f;
    unsigned int u = x.u;
    unsigned int r = u + 0x7FFFu + ((u >> 16) & 1u);   // round-to-nearest-even
    return (unsigned short)(r >> 16);
}

#if USE_ASYNC_LDS
static __device__ __forceinline__ void async_cp16(const unsigned short* gsrc,
                                                  unsigned short* ldst) {
    __builtin_amdgcn_global_load_async_to_lds_b128(
        (v4i_*)gsrc,
        (__attribute__((address_space(3))) v4i_*)ldst, 0, 0);
}
#endif

// ---------------------------------------------------------------- prep kernels
__global__ void feats_to_bf16_k(const float* __restrict__ x,
                                unsigned short* __restrict__ o, int n) {
    int i = blockIdx.x * blockDim.x + threadIdx.x;
    if (i < n) o[i] = f2bf_raw(x[i]);
}

template <int CIN>
__global__ void transpose_weights_k(const float* __restrict__ W,   // [9, CIN, 64]
                                    unsigned short* __restrict__ Wt) { // [9, 64, CIN]
    int i = blockIdx.x * blockDim.x + threadIdx.x;
    const int total = KOFF * COUTC * CIN;
    if (i >= total) return;
    int ci = i % CIN;
    int co = (i / CIN) % COUTC;
    int k  = i / (CIN * COUTC);
    Wt[i] = f2bf_raw(W[(k * CIN + ci) * COUTC + co]);
}

// -------------------------------------------------- gather + WMMA sparse conv
// Block: 256 threads = 8 wave32. Tile: 64 voxel rows x 64 cout.
// Double-buffered LDS pipeline over the 9 taps: async global->LDS staging of
// tap k+1 (A tile + weight tile) overlaps the WMMAs of tap k.
template <int CIN>
__global__ __launch_bounds__(256)
void conv_wmma_k(const unsigned short* __restrict__ act,  // [N, CIN] bf16
                 const int*            __restrict__ nbr,  // [9, N]
                 const unsigned short* __restrict__ Wt,   // [9, 64, CIN] bf16 (transposed)
                 float*                __restrict__ y)    // [N, 64] f32
{
    __shared__ __align__(16) unsigned short sA[2][64 * CIN];
    __shared__ __align__(16) unsigned short sW[2][COUTC * CIN];

    const int tid  = threadIdx.x;
    const int lane = tid & 31;
    const int wv   = tid >> 5;
    const int rt   = wv & 3;
    const int cb   = (wv >> 2) * 32;
    const int rowbase = blockIdx.x * 64;

    const int r    = tid >> 2;          // staging row 0..63 (4 threads/row)
    const int part = tid & 3;
    const int n0   = rowbase + r;
    const int WPT  = (COUTC * CIN) / 256;   // weight elems per thread (8 / 16)

    // hoist all 9 neighbor indices (coalesced, independent loads)
    int idxs[KOFF];
#pragma unroll
    for (int k = 0; k < KOFF; ++k) idxs[k] = nbr[k * N_VOX + n0];

    v8f acc0 = {};
    v8f acc1 = {};

    // stage (A tile row chunk + weight tile chunk) for tap k into buffer buf.
    // Always issues exactly 2*(CIN/32) async b128 per thread (deterministic
    // ASYNCcnt); invalid rows are clamped to row 0 and zero-patched later.
    auto stage = [&](int buf, int k) {
        const int ic = idxs[k] < 0 ? 0 : idxs[k];
        const unsigned short* asrc = &act[(size_t)ic * CIN + part * (CIN / 4)];
        unsigned short*       adst = &sA[buf][r * CIN + part * (CIN / 4)];
        const unsigned short* wsrc = &Wt[(size_t)k * COUTC * CIN + tid * WPT];
        unsigned short*       wdst = &sW[buf][tid * WPT];
#if USE_ASYNC_LDS
#pragma unroll
        for (int j = 0; j < CIN / 32; ++j) async_cp16(asrc + 8 * j, adst + 8 * j);
#pragma unroll
        for (int j = 0; j < CIN / 32; ++j) async_cp16(wsrc + 8 * j, wdst + 8 * j);
#else
        uint4 ta[CIN / 32], tw[CIN / 32];
#pragma unroll
        for (int j = 0; j < CIN / 32; ++j) ta[j] = ((const uint4*)asrc)[j];
#pragma unroll
        for (int j = 0; j < CIN / 32; ++j) tw[j] = ((const uint4*)wsrc)[j];
#pragma unroll
        for (int j = 0; j < CIN / 32; ++j) ((uint4*)adst)[j] = ta[j];
#pragma unroll
        for (int j = 0; j < CIN / 32; ++j) ((uint4*)wdst)[j] = tw[j];
#endif
    };

    stage(0, 0);   // prologue: tap 0 into buffer 0

#pragma unroll
    for (int k = 0; k < KOFF; ++k) {
        const int cur = k & 1;
        if (k + 1 < KOFF) {
            stage(cur ^ 1, k + 1);                       // overlap next tap
#if USE_ASYNC_LDS
            __builtin_amdgcn_s_wait_asynccnt(2 * (CIN / 32));  // batch k done
#endif
        } else {
#if USE_ASYNC_LDS
            __builtin_amdgcn_s_wait_asynccnt(0);
#endif
        }
        // zero-patch rows whose neighbor is masked out
        if (idxs[k] < 0) {
            uint4* dz = (uint4*)&sA[cur][r * CIN + part * (CIN / 4)];
#pragma unroll
            for (int j = 0; j < CIN / 32; ++j) dz[j] = make_uint4(0, 0, 0, 0);
        }
        __syncthreads();

#pragma unroll
        for (int cc = 0; cc < CIN / 32; ++cc) {
            // A fragment: lane<16 -> K0..15; lane>=16 -> K16..31
            const int klo  = (lane < 16 ? 0 : 16) + cc * 32;
            const int arow = 16 * rt + (lane & 15);
            AFrag a;
            a.h[0] = *(const v8bf*)&sA[cur][arow * CIN + klo];
            a.h[1] = *(const v8bf*)&sA[cur][arow * CIN + klo + 8];

            // B fragment: lane<16 -> N=lane, K0..15; lane>=16 -> N=lane-16, K16..31
            const int kw = (lane < 16 ? 0 : 16) + cc * 32;
            const int c0 = cb + (lane & 15);
            AFrag b0, b1;
            b0.h[0] = *(const v8bf*)&sW[cur][c0 * CIN + kw];
            b0.h[1] = *(const v8bf*)&sW[cur][c0 * CIN + kw + 8];
            b1.h[0] = *(const v8bf*)&sW[cur][(c0 + 16) * CIN + kw];
            b1.h[1] = *(const v8bf*)&sW[cur][(c0 + 16) * CIN + kw + 8];

            acc0 = __builtin_amdgcn_wmma_f32_16x16x32_bf16(
                false, a.v, false, b0.v, (short)0, acc0, false, false);
            acc1 = __builtin_amdgcn_wmma_f32_16x16x32_bf16(
                false, a.v, false, b1.v, (short)0, acc1, false, false);
        }
        __syncthreads();
    }

    // ---- epilogue: C/D layout VGPR i -> M=i (lanes 0-15), M=8+i (lanes 16-31)
    const int rA = rowbase + 16 * rt + (lane < 16 ? 0 : 8);
    const int cA = cb + (lane & 15);
#pragma unroll
    for (int i = 0; i < 8; ++i) {
        y[(size_t)(rA + i) * COUTC + cA]      = acc0[i];
        y[(size_t)(rA + i) * COUTC + cA + 16] = acc1[i];
    }
}

// ------------------------------------------------- deterministic BN statistics
template <bool LRELU>
__global__ __launch_bounds__(256)
void stats_partial_k(const float* __restrict__ y, float* __restrict__ part) {
    __shared__ float red[256];
    const int tid = threadIdx.x;
    const int c   = tid & 63;
    const int rowStart = blockIdx.x * 4 + (tid >> 6);
    float s = 0.f, s2 = 0.f;
    for (int r = rowStart; r < N_VOX; r += 1024) {
        float v = y[(size_t)r * 64 + c];
        if (LRELU) v = v > 0.f ? v : 0.01f * v;
        s += v; s2 += v * v;
    }
    red[tid] = s;  __syncthreads();
    if (tid < 64)
        part[blockIdx.x * 128 + c] =
            red[tid] + red[tid + 64] + red[tid + 128] + red[tid + 192];
    __syncthreads();
    red[tid] = s2; __syncthreads();
    if (tid < 64)
        part[blockIdx.x * 128 + 64 + c] =
            red[tid] + red[tid + 64] + red[tid + 128] + red[tid + 192];
}

__global__ void stats_final_k(const float* __restrict__ part, float* __restrict__ stats) {
    int j = threadIdx.x;  // 0..127  ({sum[64], sumsq[64]})
    float s = 0.f;
    for (int b = 0; b < 256; ++b) s += part[b * 128 + j];
    stats[j] = s;
}

// ---------------------------------------------- lrelu+BN -> bf16 activations
__global__ void bn_lrelu_to_bf16_k(const float* __restrict__ y,
                                   const float* __restrict__ stats,
                                   const float* __restrict__ g,
                                   const float* __restrict__ b,
                                   unsigned short* __restrict__ o) {
    int i = blockIdx.x * blockDim.x + threadIdx.x;
    if (i >= N_VOX * 64) return;
    int c = i & 63;
    float m   = stats[c] * (1.f / N_VOX);
    float var = stats[64 + c] * (1.f / N_VOX) - m * m;
    float v = y[i];
    v = v > 0.f ? v : 0.01f * v;
    o[i] = f2bf_raw(g[c] * (v - m) * rsqrtf(var + BN_EPS) + b[c]);
}

// ----------------------------------------------------- final BN + BN + add
__global__ void final_add_k(const float* __restrict__ ya, const float* __restrict__ sa,
                            const float* __restrict__ ga, const float* __restrict__ ba,
                            const float* __restrict__ yb, const float* __restrict__ sb,
                            const float* __restrict__ gb, const float* __restrict__ bb,
                            float* __restrict__ out) {
    int i = blockIdx.x * blockDim.x + threadIdx.x;
    if (i >= N_VOX * 64) return;
    int c = i & 63;
    float ma = sa[c] * (1.f / N_VOX);
    float va = sa[64 + c] * (1.f / N_VOX) - ma * ma;
    float xa = ga[c] * (ya[i] - ma) * rsqrtf(va + BN_EPS) + ba[c];
    float mb = sb[c] * (1.f / N_VOX);
    float vb = sb[64 + c] * (1.f / N_VOX) - mb * mb;
    float xb = gb[c] * (yb[i] - mb) * rsqrtf(vb + BN_EPS) + bb[c];
    out[i] = xa + xb;
}

// -------------------------------------------------------------------- launch
extern "C" void kernel_launch(void* const* d_in, const int* in_sizes, int n_in,
                              void* d_out, int out_size, void* d_ws, size_t ws_size,
                              hipStream_t stream) {
    const float* feats = (const float*)d_in[0];
    const int*   nbr_a = (const int*)d_in[1];
    const int*   nbr_b = (const int*)d_in[2];
    const float* W1    = (const float*)d_in[3];
    const float* W12   = (const float*)d_in[4];
    const float* W2    = (const float*)d_in[5];
    const float* W3    = (const float*)d_in[6];
    const float* g0  = (const float*)d_in[7];  const float* b0  = (const float*)d_in[8];
    const float* g02 = (const float*)d_in[9];  const float* b02 = (const float*)d_in[10];
    const float* g1  = (const float*)d_in[11]; const float* b1  = (const float*)d_in[12];
    const float* g2  = (const float*)d_in[13]; const float* b2  = (const float*)d_in[14];

    char* ws = (char*)d_ws;
    size_t off = 0;
    auto alloc = [&](size_t bytes) -> void* {
        void* p = ws + off;
        off = (off + bytes + 255) & ~(size_t)255;
        return p;
    };
    unsigned short* featsb = (unsigned short*)alloc((size_t)N_VOX * 32 * 2);
    unsigned short* W1t    = (unsigned short*)alloc((size_t)KOFF * 64 * 32 * 2);
    unsigned short* W2t    = (unsigned short*)alloc((size_t)KOFF * 64 * 32 * 2);
    unsigned short* W12t   = (unsigned short*)alloc((size_t)KOFF * 64 * 64 * 2);
    unsigned short* W3t    = (unsigned short*)alloc((size_t)KOFF * 64 * 64 * 2);
    float* sc0 = (float*)alloc((size_t)N_VOX * 64 * 4);
    float* r0  = (float*)alloc((size_t)N_VOX * 64 * 4);
    unsigned short* sc0n = (unsigned short*)alloc((size_t)N_VOX * 64 * 2);
    unsigned short* r0n  = (unsigned short*)alloc((size_t)N_VOX * 64 * 2);
    float* part  = (float*)alloc(256 * 128 * 4);
    float* stats = (float*)alloc(4 * 128 * 4);
    float* sc1 = sc0;   // sc0 dead after bn_lrelu_to_bf16
    float* r1  = r0;

    const int EW = 256;
    // prep: bf16 activations + transposed bf16 weights
    feats_to_bf16_k<<<(N_VOX * 32 + EW - 1) / EW, EW, 0, stream>>>(feats, featsb, N_VOX * 32);
    transpose_weights_k<32><<<(KOFF * 64 * 32 + EW - 1) / EW, EW, 0, stream>>>(W1, W1t);
    transpose_weights_k<32><<<(KOFF * 64 * 32 + EW - 1) / EW, EW, 0, stream>>>(W2, W2t);
    transpose_weights_k<64><<<(KOFF * 64 * 64 + EW - 1) / EW, EW, 0, stream>>>(W12, W12t);
    transpose_weights_k<64><<<(KOFF * 64 * 64 + EW - 1) / EW, EW, 0, stream>>>(W3, W3t);

    const int CONV_BLOCKS = N_VOX / 64;  // 3125, exact
    // first convs (Cin=32)
    conv_wmma_k<32><<<CONV_BLOCKS, 256, 0, stream>>>(featsb, nbr_a, W1t, sc0);
    conv_wmma_k<32><<<CONV_BLOCKS, 256, 0, stream>>>(featsb, nbr_b, W2t, r0);

    // BN stats of lrelu(conv) then normalize to bf16
    stats_partial_k<true><<<256, 256, 0, stream>>>(sc0, part);
    stats_final_k<<<1, 128, 0, stream>>>(part, stats + 0);
    stats_partial_k<true><<<256, 256, 0, stream>>>(r0, part);
    stats_final_k<<<1, 128, 0, stream>>>(part, stats + 128);

    const int NE = N_VOX * 64;
    bn_lrelu_to_bf16_k<<<(NE + EW - 1) / EW, EW, 0, stream>>>(sc0, stats + 0,   g0, b0, sc0n);
    bn_lrelu_to_bf16_k<<<(NE + EW - 1) / EW, EW, 0, stream>>>(r0,  stats + 128, g1, b1, r0n);

    // second convs (Cin=64)
    conv_wmma_k<64><<<CONV_BLOCKS, 256, 0, stream>>>(sc0n, nbr_b, W12t, sc1);
    conv_wmma_k<64><<<CONV_BLOCKS, 256, 0, stream>>>(r0n,  nbr_a, W3t,  r1);

    // BN stats (no lrelu) + fused final add
    stats_partial_k<false><<<256, 256, 0, stream>>>(sc1, part);
    stats_final_k<<<1, 128, 0, stream>>>(part, stats + 256);
    stats_partial_k<false><<<256, 256, 0, stream>>>(r1, part);
    stats_final_k<<<1, 128, 0, stream>>>(part, stats + 384);

    final_add_k<<<(NE + EW - 1) / EW, EW, 0, stream>>>(
        sc1, stats + 256, g02, b02,
        r1,  stats + 384, g2,  b2, (float*)d_out);
}
